// LSTMAE_80685255622883
// MI455X (gfx1250) — compile-verified, matrix operand order checked by reading, not run on hardware
//
#include <hip/hip_runtime.h>
#include <hip/hip_bf16.h>
#include <cstdint>

// ---------------------------------------------------------------------------
// LSTM autoencoder for MI455X (gfx1250, wave32, WMMA).
//   B=32, T=512, D_IN=128, H=256 (encoder), H=128 (decoder), LATENT=512.
// Strategy:
//   * All matmuls in bf16 WMMA (v_wmma_f32_16x16x32_bf16), fp32 accumulate.
//   * Input projections = big parallel GEMMs (M = T*B = 16384).
//   * Recurrence = persistent workgroup per direction; h double-buffered in
//     LDS (ds_load_b128 A-fragments), Whh streamed from L2, c in registers.
//   * Gate math (sigmoid/tanh) fp32 in registers; one barrier per timestep.
// ---------------------------------------------------------------------------

#define TSEQ  512
#define BATCH 32
#define DIN   128

typedef __attribute__((ext_vector_type(16))) __bf16 v16bf;
typedef __attribute__((ext_vector_type(8)))  float  f32x8;

union FragB16 {
  v16bf v;
  uint4 q[2];   // two 16-byte halves -> global_load_b128 / ds_load_b128
};

__device__ __forceinline__ unsigned short f2b(float f) {
  union { float f; unsigned int u; } x; x.f = f;
  unsigned int r = x.u + 0x7FFFu + ((x.u >> 16) & 1u);   // round-nearest-even
  return (unsigned short)(r >> 16);
}
__device__ __forceinline__ float b2f(unsigned short b) {
  union { unsigned int u; float f; } x; x.u = ((unsigned int)b) << 16;
  return x.f;
}
__device__ __forceinline__ float sigm(float x) { return 1.0f / (1.0f + __expf(-x)); }

__device__ __forceinline__ f32x8 wmma_bf16(const FragB16& a, const FragB16& b, f32x8 c) {
  // D = A(16x32) * B(32x16) + C ; 8 args: neg_a, A, neg_b, B, c_mod, C, reuse_a, reuse_b
  return __builtin_amdgcn_wmma_f32_16x16x32_bf16(false, a.v, false, b.v, (short)0, c,
                                                 false, false);
}

// ---------------------------------------------------------------------------
// x [B,T,DIN] f32  ->  xT [(t*B+b), DIN] bf16
// ---------------------------------------------------------------------------
__global__ void __launch_bounds__(256)
xpose_kernel(const float* __restrict__ x, unsigned short* __restrict__ xT) {
  size_t i = (size_t)blockIdx.x * 256 + threadIdx.x;      // over T*B*DIN
  if (i >= (size_t)TSEQ * BATCH * DIN) return;
  int    d  = (int)(i & (DIN - 1));
  size_t tb = i >> 7;                                     // DIN == 128
  int    b  = (int)(tb & (BATCH - 1));
  int    t  = (int)(tb >> 5);                             // BATCH == 32
  xT[i] = f2b(x[((size_t)b * TSEQ + t) * DIN + d]);
}

__global__ void __launch_bounds__(256)
f32_to_bf16_kernel(const float* __restrict__ src, unsigned short* __restrict__ dst, int n) {
  int i = blockIdx.x * 256 + threadIdx.x;
  if (i < n) dst[i] = f2b(src[i]);
}

// ---------------------------------------------------------------------------
// GEMM: out[m,n] = sum_k A[m,k] * W[n,k]  (+ bias[n])
//   A bf16 [M,K] row-major, W bf16 [N,K] row-major.
//   Block: 256 threads = 8 waves; each wave one 16-row M-tile x 64 cols.
//   B tile (64x32 bf16) staged in LDS, shared by all 8 waves.
//   outMode 0: bf16 store [M,N]; outMode 1: f32 store with [t*B+b] -> [b,t].
// ---------------------------------------------------------------------------
__global__ void __launch_bounds__(256)
gemm_bf16_wmma(const unsigned short* __restrict__ A,
               const unsigned short* __restrict__ W,
               void* __restrict__ out, const float* __restrict__ bias,
               int N, int K, int outMode) {
  const int tid  = threadIdx.x;
  const int wave = tid >> 5, lane = tid & 31;
  const int ln = lane & 15, hi = lane >> 4;
  const int m0    = (blockIdx.x * 8 + wave) * 16;
  const int nbase = blockIdx.y * 64;

  __shared__ unsigned short bt[64][40];   // stride 40 (80B) kills bank conflicts

  const f32x8 zf = {0.f,0.f,0.f,0.f,0.f,0.f,0.f,0.f};
  f32x8 acc[4] = {zf, zf, zf, zf};

  const int sn = tid >> 2;                // 0..63 staged column
  const int sk = (tid & 3) * 8;           // 0,8,16,24
  const unsigned short* wrow = W + (size_t)(nbase + sn) * K + sk;

  for (int k0 = 0; k0 < K; k0 += 32) {
    __syncthreads();                      // previous tile fully consumed
    *reinterpret_cast<uint4*>(&bt[sn][sk]) =
        *reinterpret_cast<const uint4*>(wrow + k0);
    __syncthreads();

    FragB16 a;
    const unsigned short* ap = A + (size_t)(m0 + ln) * K + k0 + 8 * hi;
    a.q[0] = *reinterpret_cast<const uint4*>(ap);
    a.q[1] = *reinterpret_cast<const uint4*>(ap + 16);

#pragma unroll
    for (int j = 0; j < 4; ++j) {
      FragB16 b;
      const unsigned short* bp = &bt[j * 16 + ln][8 * hi];
      b.q[0] = *reinterpret_cast<const uint4*>(bp);
      b.q[1] = *reinterpret_cast<const uint4*>(bp + 16);
      acc[j] = wmma_bf16(a, b, acc[j]);
    }
  }

#pragma unroll
  for (int j = 0; j < 4; ++j) {
    const int col = nbase + j * 16 + ln;
    const float bv = bias ? bias[col] : 0.0f;
#pragma unroll
    for (int r = 0; r < 8; ++r) {
      const int row = m0 + hi * 8 + r;
      const float v = acc[j][r] + bv;
      if (outMode == 0) {
        reinterpret_cast<unsigned short*>(out)[(size_t)row * N + col] = f2b(v);
      } else {
        const int bb = row & (BATCH - 1);
        const int tt = row >> 5;
        reinterpret_cast<float*>(out)[((size_t)bb * TSEQ + tt) * N + col] = v;
      }
    }
  }
}

// ---------------------------------------------------------------------------
// Persistent LSTM scan. One block per direction (blockIdx.x = dir).
//   HD = hidden dim; block = 32*(HD/16) threads, wave w owns j-tile w
//   (columns j..j+15) for ALL FOUR GATES -> cell update is register-local.
//   g = xg[t] + bias + h_{t-1} @ Whh^T ; h in LDS (bf16, double-buffered).
//   xg bf16 [T*B, 4H] gate order i,f,g,o; Whh bf16 [4H, HD].
//   hout bf16 [(t*B+b), outLd], this direction writes cols [dir*HD, dir*HD+HD).
// ---------------------------------------------------------------------------
template <int HD>
__global__ void __launch_bounds__(512)
lstm_scan_kernel(const unsigned short* __restrict__ xg0,
                 const unsigned short* __restrict__ xg1,
                 const unsigned short* __restrict__ whh0,
                 const unsigned short* __restrict__ whh1,
                 const float* __restrict__ bias0, const float* __restrict__ bias1,
                 unsigned short* __restrict__ hout, int outLd) {
  constexpr int FOURH = 4 * HD;
  const int dir = blockIdx.x;
  const unsigned short* __restrict__ xg  = dir ? xg1  : xg0;
  const unsigned short* __restrict__ whh = dir ? whh1 : whh0;
  const float* __restrict__ bias         = dir ? bias1 : bias0;
  const int colOff = dir * HD;

  const int tid  = threadIdx.x;
  const int wave = tid >> 5, lane = tid & 31;
  const int ln = lane & 15, hi = lane >> 4;
  const int jt = wave;                       // one hidden j-tile per wave

  __shared__ unsigned short hbuf[2][BATCH][HD];
  for (int i = tid; i < 2 * BATCH * HD; i += blockDim.x)
    reinterpret_cast<unsigned short*>(hbuf)[i] = 0;
  __syncthreads();

  float bg[4];
#pragma unroll
  for (int g = 0; g < 4; ++g) bg[g] = bias[g * HD + jt * 16 + ln];

  float creg[2][8];
#pragma unroll
  for (int mt = 0; mt < 2; ++mt)
#pragma unroll
    for (int r = 0; r < 8; ++r) creg[mt][r] = 0.0f;

  const f32x8 zf = {0.f,0.f,0.f,0.f,0.f,0.f,0.f,0.f};

  for (int t = 0; t < TSEQ; ++t) {
    const int tt  = dir ? (TSEQ - 1 - t) : t;
    const int cur = t & 1, nxt = cur ^ 1;

    f32x8 acc[4][2];
#pragma unroll
    for (int g = 0; g < 4; ++g) { acc[g][0] = zf; acc[g][1] = zf; }

    for (int k0 = 0; k0 < HD; k0 += 32) {
      FragB16 a[2];                          // h_{t-1} fragments from LDS
#pragma unroll
      for (int mt = 0; mt < 2; ++mt) {
        const unsigned short* hp = &hbuf[cur][mt * 16 + ln][k0 + 8 * hi];
        a[mt].q[0] = *reinterpret_cast<const uint4*>(hp);
        a[mt].q[1] = *reinterpret_cast<const uint4*>(hp + 16);
      }
#pragma unroll
      for (int g = 0; g < 4; ++g) {
        FragB16 b;                           // Whh fragments from L2
        const unsigned short* wp =
            whh + (size_t)(g * HD + jt * 16 + ln) * HD + k0 + 8 * hi;
        b.q[0] = *reinterpret_cast<const uint4*>(wp);
        b.q[1] = *reinterpret_cast<const uint4*>(wp + 16);
        if (k0 + 32 < HD) __builtin_prefetch(wp + 32, 0, 0);  // global_prefetch_b8
        acc[g][0] = wmma_bf16(a[0], b, acc[g][0]);
        acc[g][1] = wmma_bf16(a[1], b, acc[g][1]);
      }
    }

    const int j = jt * 16 + ln;
#pragma unroll
    for (int mt = 0; mt < 2; ++mt) {
#pragma unroll
      for (int r = 0; r < 8; ++r) {
        const int brow = mt * 16 + hi * 8 + r;        // batch row, C/D layout
        const unsigned short* xr = xg + ((size_t)tt * BATCH + brow) * FOURH + j;
        float gi = acc[0][mt][r] + b2f(xr[0])      + bg[0];
        float gf = acc[1][mt][r] + b2f(xr[HD])     + bg[1];
        float gc = acc[2][mt][r] + b2f(xr[2 * HD]) + bg[2];
        float go = acc[3][mt][r] + b2f(xr[3 * HD]) + bg[3];
        gi = sigm(gi); gf = sigm(gf); go = sigm(go); gc = tanhf(gc);
        float c = gf * creg[mt][r] + gi * gc;
        creg[mt][r] = c;
        const unsigned short hb = f2b(go * tanhf(c));
        hbuf[nxt][brow][j] = hb;
        hout[((size_t)tt * BATCH + brow) * outLd + colOff + j] = hb;
      }
    }
    __syncthreads();   // new h visible before next step; old h fully consumed
  }
}

// ---------------------------------------------------------------------------
// Host launcher
// ---------------------------------------------------------------------------
extern "C" void kernel_launch(void* const* d_in, const int* in_sizes, int n_in,
                              void* d_out, int out_size, void* d_ws, size_t ws_size,
                              hipStream_t stream) {
  const float* x       = (const float*)d_in[0];
  const float* e0f_Wih = (const float*)d_in[1];
  const float* e0f_Whh = (const float*)d_in[2];
  const float* e0f_b   = (const float*)d_in[3];
  const float* e0b_Wih = (const float*)d_in[4];
  const float* e0b_Whh = (const float*)d_in[5];
  const float* e0b_b   = (const float*)d_in[6];
  const float* e1f_Wih = (const float*)d_in[7];
  const float* e1f_Whh = (const float*)d_in[8];
  const float* e1f_b   = (const float*)d_in[9];
  const float* e1b_Wih = (const float*)d_in[10];
  const float* e1b_Whh = (const float*)d_in[11];
  const float* e1b_b   = (const float*)d_in[12];
  const float* dec_Wih = (const float*)d_in[13];
  const float* dec_Whh = (const float*)d_in[14];
  const float* dec_b   = (const float*)d_in[15];
  const float* out_W   = (const float*)d_in[16];
  const float* out_b   = (const float*)d_in[17];

  const size_t M = (size_t)TSEQ * BATCH;   // 16384 rows

  char*  base = (char*)d_ws;
  size_t off  = 0;
  auto alloc = [&](size_t elems) -> unsigned short* {
    unsigned short* r = (unsigned short*)(base + off);
    off += (elems * 2 + 255) & ~(size_t)255;
    return r;
  };

  unsigned short* xT      = alloc(M * DIN);
  unsigned short* e0fWihB = alloc(1024 * 128);
  unsigned short* e0fWhhB = alloc(1024 * 256);
  unsigned short* e0bWihB = alloc(1024 * 128);
  unsigned short* e0bWhhB = alloc(1024 * 256);
  unsigned short* e1fWihB = alloc(1024 * 512);
  unsigned short* e1fWhhB = alloc(1024 * 256);
  unsigned short* e1bWihB = alloc(1024 * 512);
  unsigned short* e1bWhhB = alloc(1024 * 256);
  unsigned short* decWihB = alloc(512 * 512);
  unsigned short* decWhhB = alloc(512 * 128);
  unsigned short* outWB   = alloc(128 * 128);
  unsigned short* xgA     = alloc(M * 1024);    // fwd gate preacts (reused/layer)
  unsigned short* xgB     = alloc(M * 1024);    // bwd gate preacts
  unsigned short* h0      = alloc(M * 512);     // encoder layer0 output [fwd|bwd]
  unsigned short* h1      = alloc(M * 512);     // encoder layer1 output
  unsigned short* y       = alloc(M * 128);     // decoder hidden states

  // --- transpose x and convert weights to bf16 (recomputed each call) -----
  xpose_kernel<<<(unsigned)((M * DIN + 255) / 256), 256, 0, stream>>>(x, xT);

  struct CvtJob { const float* src; unsigned short* dst; int n; };
  const CvtJob jobs[11] = {
      {e0f_Wih, e0fWihB, 1024 * 128}, {e0f_Whh, e0fWhhB, 1024 * 256},
      {e0b_Wih, e0bWihB, 1024 * 128}, {e0b_Whh, e0bWhhB, 1024 * 256},
      {e1f_Wih, e1fWihB, 1024 * 512}, {e1f_Whh, e1fWhhB, 1024 * 256},
      {e1b_Wih, e1bWihB, 1024 * 512}, {e1b_Whh, e1bWhhB, 1024 * 256},
      {dec_Wih, decWihB, 512 * 512},  {dec_Whh, decWhhB, 512 * 128},
      {out_W,   outWB,   128 * 128}};
  for (int i = 0; i < 11; ++i)
    f32_to_bf16_kernel<<<(jobs[i].n + 255) / 256, 256, 0, stream>>>(
        jobs[i].src, jobs[i].dst, jobs[i].n);

  const dim3 gblk(256);

  // --- encoder layer 0: input projections + bidirectional scan ------------
  gemm_bf16_wmma<<<dim3(128, 16), gblk, 0, stream>>>(xT, e0fWihB, xgA, nullptr, 1024, 128, 0);
  gemm_bf16_wmma<<<dim3(128, 16), gblk, 0, stream>>>(xT, e0bWihB, xgB, nullptr, 1024, 128, 0);
  lstm_scan_kernel<256><<<2, 512, 0, stream>>>(xgA, xgB, e0fWhhB, e0bWhhB,
                                               e0f_b, e0b_b, h0, 512);

  // --- encoder layer 1 ----------------------------------------------------
  gemm_bf16_wmma<<<dim3(128, 16), gblk, 0, stream>>>(h0, e1fWihB, xgA, nullptr, 1024, 512, 0);
  gemm_bf16_wmma<<<dim3(128, 16), gblk, 0, stream>>>(h0, e1bWihB, xgB, nullptr, 1024, 512, 0);
  lstm_scan_kernel<256><<<2, 512, 0, stream>>>(xgA, xgB, e1fWhhB, e1bWhhB,
                                               e1f_b, e1b_b, h1, 512);

  // --- decoder (unidirectional, H = 128) ----------------------------------
  gemm_bf16_wmma<<<dim3(128, 8), gblk, 0, stream>>>(h1, decWihB, xgA, nullptr, 512, 512, 0);
  lstm_scan_kernel<128><<<1, 256, 0, stream>>>(xgA, xgA, decWhhB, decWhhB,
                                               dec_b, dec_b, y, 128);

  // --- output linear, fp32, fused [t*B+b] -> [b,t] transpose --------------
  gemm_bf16_wmma<<<dim3(128, 2), gblk, 0, stream>>>(y, outWB, d_out, out_b, 128, 128, 1);
}
